// PolicyNet_44513041056059
// MI455X (gfx1250) — matrix operand order repaired and checked by reading, not compiled, not run on hardware
//
#include <hip/hip_runtime.h>
#include <hip/hip_bf16.h>
#include <math.h>

// Problem dims (fixed by reference)
#define T_DIM   128
#define B_DIM   64
#define OBS_DIM 2048
#define H_DIM   1024
#define A_DIM   6
#define TB_DIM  (T_DIM * B_DIM)   // 8192

typedef __bf16 bf16_t;
typedef __attribute__((ext_vector_type(16))) __bf16 v16bf;
typedef __attribute__((ext_vector_type(8)))  __bf16 v8bf;
typedef __attribute__((ext_vector_type(8)))  float  v8f;

// ---------------------------------------------------------------------------
// Fragment loaders for V_WMMA_F32_16X16X32_BF16 (wave32).
// A (16x32, MxK) per-lane layout (ISA 7.12.2):
//   lanes 0-15  : row M = lane,    K = {kb+0..7,  kb+16..23}
//   lanes 16-31 : row M = lane-16, K = {kb+8..15, kb+24..31}
// -> two contiguous 16B loads per lane from row-major A.
// B (32x16, KxN) per-lane layout: N = lane&15 striped across lanes,
//   lanes 0-15 hold K={kb+0..7, kb+16..23}, lanes 16-31 hold K={kb+8..15, kb+24..31}
// -> two 8-bf16 loads from row n of W (stored (N,K) row-major), i.e. B=W^T.
// ---------------------------------------------------------------------------

__device__ __forceinline__ v16bf load_a_frag(const bf16_t* __restrict__ A, int lda,
                                             int m0, int kb, int lane) {
  const int g = lane >> 4, r = lane & 15;
  const bf16_t* p = A + (size_t)(m0 + r) * lda + (kb + 8 * g);
  v8bf lo = *(const v8bf*)p;
  v8bf hi = *(const v8bf*)(p + 16);
  v16bf f;
#pragma unroll
  for (int i = 0; i < 8; i++) { f[i] = lo[i]; f[8 + i] = hi[i]; }
  return f;
}

// f32 source variant (obs): convert in-register, avoids a bf16 staging pass.
__device__ __forceinline__ v16bf load_a_frag(const float* __restrict__ A, int lda,
                                             int m0, int kb, int lane) {
  const int g = lane >> 4, r = lane & 15;
  const float* p = A + (size_t)(m0 + r) * lda + (kb + 8 * g);
  v8f lo = *(const v8f*)p;
  v8f hi = *(const v8f*)(p + 16);
  v16bf f;
#pragma unroll
  for (int i = 0; i < 8; i++) { f[i] = (__bf16)lo[i]; f[8 + i] = (__bf16)hi[i]; }
  return f;
}

__device__ __forceinline__ v16bf load_b_frag(const bf16_t* __restrict__ W, int ldw,
                                             int n0, int kb, int lane) {
  const int g = lane >> 4, nn = lane & 15;
  const bf16_t* p = W + (size_t)(n0 + nn) * ldw + (kb + 8 * g);
  v8bf lo = *(const v8bf*)p;
  v8bf hi = *(const v8bf*)(p + 16);
  v16bf f;
#pragma unroll
  for (int i = 0; i < 8; i++) { f[i] = lo[i]; f[8 + i] = hi[i]; }
  return f;
}

// ---------------------------------------------------------------------------
// Generic WMMA GEMM:  C = act( A1 @ B1^T  [+ A2m @ B2^T]  + bias1 [+ bias2] )
//   A1: (M x K1) row-major (TA = bf16 or float)
//   B1: (N x K1) row-major bf16 (i.e. weight matrix, used transposed)
//   A2: optional (M x K2) bf16; rows with mask2[row]!=0 are zeroed (done-mask,
//       branch-free per-lane cndmask select)
//   Output: f32 (WF) and/or bf16 (WB), leading dim N.
// Block: 256 threads = 8 waves arranged 2(M) x 4(N).
// Wave tile: (IM*16) x (IN*16).  Each K-step: IM+IN frag loads, IM*IN WMMAs.
// B operand gets an unconditional WGP-scope prefetch 512B ahead every other
// K-step (buffers live mid-workspace, so the small overrun is valid memory;
// speculative prefetch drops on translation failure anyway).
// ---------------------------------------------------------------------------
template <typename TA, int IM, int IN, bool RELU, bool WF, bool WB>
__global__ __launch_bounds__(256) void wmma_gemm_kernel(
    const TA* __restrict__ A1, const bf16_t* __restrict__ B1, int K1,
    const bf16_t* __restrict__ A2, const bf16_t* __restrict__ B2, int K2,
    const int* __restrict__ mask2,
    const float* __restrict__ bias1, const float* __restrict__ bias2,
    float* __restrict__ Cf, bf16_t* __restrict__ Cb, int N) {
  const int lane = threadIdx.x & 31;   // gfx1250: wave32
  const int wave = threadIdx.x >> 5;
  const int m0 = blockIdx.y * (2 * IM * 16) + (wave >> 2) * (IM * 16);
  const int n0 = blockIdx.x * (4 * IN * 16) + (wave & 3) * (IN * 16);
  const int nn_l = lane & 15, g_l = lane >> 4;

  v8f acc[IM][IN];
#pragma unroll
  for (int i = 0; i < IM; i++)
#pragma unroll
    for (int j = 0; j < IN; j++)
#pragma unroll
      for (int e = 0; e < 8; e++) acc[i][j][e] = 0.0f;

  // Pass 1: A1 @ B1^T
#pragma unroll 2
  for (int k = 0; k < K1; k += 32) {
    if ((k & 32) == 0) {   // compile-time per unrolled copy
#pragma unroll
      for (int j = 0; j < IN; j++)
        __builtin_prefetch(B1 + (size_t)(n0 + 16 * j + nn_l) * K1 +
                               (k + 256 + 16 * g_l), 0, 3);
    }
    v16bf a[IM], b[IN];
#pragma unroll
    for (int i = 0; i < IM; i++) a[i] = load_a_frag(A1, K1, m0 + 16 * i, k, lane);
#pragma unroll
    for (int j = 0; j < IN; j++) b[j] = load_b_frag(B1, K1, n0 + 16 * j, k, lane);
#pragma unroll
    for (int i = 0; i < IM; i++)
#pragma unroll
      for (int j = 0; j < IN; j++)
        acc[i][j] = __builtin_amdgcn_wmma_f32_16x16x32_bf16(
            false, a[i], false, b[j], (short)0, acc[i][j], false, false);
  }

  // Pass 2 (optional): masked A2 @ B2^T accumulated into same tiles
  if (A2 != nullptr) {
#pragma unroll 2
    for (int k = 0; k < K2; k += 32) {
      if ((k & 32) == 0) {
#pragma unroll
        for (int j = 0; j < IN; j++)
          __builtin_prefetch(B2 + (size_t)(n0 + 16 * j + nn_l) * K2 +
                                 (k + 256 + 16 * g_l), 0, 3);
      }
      v16bf a[IM], b[IN];
#pragma unroll
      for (int i = 0; i < IM; i++) {
        a[i] = load_a_frag(A2, K2, m0 + 16 * i, k, lane);
        if (mask2 != nullptr) {
          // branch-free per-lane row zeroing (done -> h row contributes 0)
          const bool z = (mask2[m0 + 16 * i + nn_l] != 0);
#pragma unroll
          for (int e = 0; e < 16; e++) a[i][e] = z ? (__bf16)0.0f : a[i][e];
        }
      }
#pragma unroll
      for (int j = 0; j < IN; j++) b[j] = load_b_frag(B2, K2, n0 + 16 * j, k, lane);
#pragma unroll
      for (int i = 0; i < IM; i++)
#pragma unroll
        for (int j = 0; j < IN; j++)
          acc[i][j] = __builtin_amdgcn_wmma_f32_16x16x32_bf16(
              false, a[i], false, b[j], (short)0, acc[i][j], false, false);
    }
  }

  // Epilogue. C/D layout: VGPR r, lane l: row = m0+8*(l>>4)+r, col = n0+(l&15)
#pragma unroll
  for (int i = 0; i < IM; i++) {
#pragma unroll
    for (int j = 0; j < IN; j++) {
      const int col = n0 + 16 * j + nn_l;
      float bv = bias1[col];
      if (bias2 != nullptr) bv += bias2[col];
#pragma unroll
      for (int r = 0; r < 8; r++) {
        const int row = m0 + 16 * i + 8 * g_l + r;
        float v = acc[i][j][r] + bv;
        if (RELU) v = fmaxf(v, 0.0f);
        const size_t idx = (size_t)row * N + col;
        if (WF) Cf[idx] = v;
        if (WB) Cb[idx] = (bf16_t)v;
      }
    }
  }
}

// ---------------------------------------------------------------------------
// Elementwise helpers / LSTM cell / heads
// ---------------------------------------------------------------------------
__global__ void cvt_f32_to_bf16(const float* __restrict__ in,
                                bf16_t* __restrict__ out, int n) {
  int i = blockIdx.x * blockDim.x + threadIdx.x;
  if (i < n) out[i] = (bf16_t)in[i];
}

__device__ __forceinline__ float sigmoidf_(float x) {
  return 1.0f / (1.0f + __expf(-x));
}

// gates (B x 4H, order i|f|g|o) already include the done-masked h matmul.
// c mask applied here.  Writes f32 h,c state, bf16 h (next GEMM operand),
// and optionally the core_output slot (layer 1 only).
__global__ void lstm_cell_kernel(const float* __restrict__ gates,
                                 const int* __restrict__ done_t,
                                 float* __restrict__ h, float* __restrict__ c,
                                 bf16_t* __restrict__ hbf,
                                 bf16_t* __restrict__ out_t) {
  int idx = blockIdx.x * blockDim.x + threadIdx.x;
  if (idx >= B_DIM * H_DIM) return;
  const int b = idx >> 10;            // /H_DIM
  const int j = idx & (H_DIM - 1);
  const float nd = (done_t[b] != 0) ? 0.0f : 1.0f;
  const float* g = gates + (size_t)b * 4 * H_DIM;
  const float ig = sigmoidf_(g[j]);
  const float fg = sigmoidf_(g[H_DIM + j]);
  const float gg = tanhf(g[2 * H_DIM + j]);
  const float og = sigmoidf_(g[3 * H_DIM + j]);
  const float cn = fg * (c[idx] * nd) + ig * gg;
  const float hn = og * tanhf(cn);
  c[idx] = cn;
  h[idx] = hn;
  hbf[idx] = (bf16_t)hn;
  if (out_t != nullptr) out_t[idx] = (bf16_t)hn;
}

// Policy/baseline heads + argmax: one wave32 per row, 7 dots of length 1024.
__global__ __launch_bounds__(256) void heads_kernel(
    const bf16_t* __restrict__ core, const float* __restrict__ Wp,
    const float* __restrict__ bp, const float* __restrict__ Wb,
    const float* __restrict__ bb, float* __restrict__ logits,
    float* __restrict__ baseline, float* __restrict__ action, int rows) {
  const int gw = (int)((blockIdx.x * blockDim.x + threadIdx.x) >> 5);
  const int lane = threadIdx.x & 31;
  if (gw >= rows) return;
  const bf16_t* x = core + (size_t)gw * H_DIM;
  float s[A_DIM + 1];
#pragma unroll
  for (int a = 0; a <= A_DIM; a++) s[a] = 0.0f;
  for (int k = lane; k < H_DIM; k += 32) {
    const float xv = (float)x[k];
#pragma unroll
    for (int a = 0; a < A_DIM; a++) s[a] += xv * Wp[a * H_DIM + k];
    s[A_DIM] += xv * Wb[k];
  }
#pragma unroll
  for (int a = 0; a <= A_DIM; a++)
    for (int off = 16; off > 0; off >>= 1) s[a] += __shfl_xor(s[a], off, 32);
  if (lane == 0) {
    float best = -1e30f; int bi = 0;
#pragma unroll
    for (int a = 0; a < A_DIM; a++) {
      const float l = s[a] + bp[a];
      logits[(size_t)gw * A_DIM + a] = l;
      if (l > best) { best = l; bi = a; }   // strict > : first max wins (argmax)
    }
    baseline[gw] = s[A_DIM] + bb[0];
    action[gw] = (float)bi;
  }
}

// ---------------------------------------------------------------------------
// Host-side orchestration (graph-capturable: only async ops on `stream`)
// ---------------------------------------------------------------------------
extern "C" void kernel_launch(void* const* d_in, const int* in_sizes, int n_in,
                              void* d_out, int out_size, void* d_ws, size_t ws_size,
                              hipStream_t stream) {
  const float* obs  = (const float*)d_in[0];
  const int*   done = (const int*)  d_in[1];
  const float* h0   = (const float*)d_in[2];
  const float* c0   = (const float*)d_in[3];
  const float* W1   = (const float*)d_in[4];
  const float* b1   = (const float*)d_in[5];
  const float* W2   = (const float*)d_in[6];
  const float* b2   = (const float*)d_in[7];
  const float* Wih0 = (const float*)d_in[8];
  const float* Whh0 = (const float*)d_in[9];
  const float* bih0 = (const float*)d_in[10];
  const float* bhh0 = (const float*)d_in[11];
  const float* Wih1 = (const float*)d_in[12];
  const float* Whh1 = (const float*)d_in[13];
  const float* bih1 = (const float*)d_in[14];
  const float* bhh1 = (const float*)d_in[15];
  const float* Wp   = (const float*)d_in[16];
  const float* bp   = (const float*)d_in[17];
  const float* Wb   = (const float*)d_in[18];
  const float* bb   = (const float*)d_in[19];
  float* out = (float*)d_out;

  // Workspace carve-up (256B aligned)
  char* w = (char*)d_ws;
  auto carve = [&](size_t bytes) -> char* {
    char* p = w; w += (bytes + 255) & ~(size_t)255; return p;
  };
  bf16_t* W1b   = (bf16_t*)carve((size_t)H_DIM * OBS_DIM * 2);
  bf16_t* W2b   = (bf16_t*)carve((size_t)H_DIM * H_DIM * 2);
  bf16_t* Wih0b = (bf16_t*)carve((size_t)4 * H_DIM * H_DIM * 2);
  bf16_t* Whh0b = (bf16_t*)carve((size_t)4 * H_DIM * H_DIM * 2);
  bf16_t* Wih1b = (bf16_t*)carve((size_t)4 * H_DIM * H_DIM * 2);
  bf16_t* Whh1b = (bf16_t*)carve((size_t)4 * H_DIM * H_DIM * 2);
  bf16_t* X1b   = (bf16_t*)carve((size_t)TB_DIM * H_DIM * 2);   // relu MLP1 out
  bf16_t* CIb   = (bf16_t*)carve((size_t)TB_DIM * H_DIM * 2);   // core_input
  bf16_t* COb   = (bf16_t*)carve((size_t)TB_DIM * H_DIM * 2);   // core_output
  float*  gates = (float*) carve((size_t)B_DIM * 4 * H_DIM * 4);
  float*  hst   = (float*) carve((size_t)2 * B_DIM * H_DIM * 4);
  float*  cst   = (float*) carve((size_t)2 * B_DIM * H_DIM * 4);
  bf16_t* hbf   = (bf16_t*)carve((size_t)2 * B_DIM * H_DIM * 2);

  // 1) Per-launch weight conversion f32 -> bf16 (deterministic, no caching)
  auto cvt = [&](const float* src, bf16_t* dst, int n) {
    cvt_f32_to_bf16<<<(n + 255) / 256, 256, 0, stream>>>(src, dst, n);
  };
  cvt(W1,   W1b,   H_DIM * OBS_DIM);
  cvt(W2,   W2b,   H_DIM * H_DIM);
  cvt(Wih0, Wih0b, 4 * H_DIM * H_DIM);
  cvt(Whh0, Whh0b, 4 * H_DIM * H_DIM);
  cvt(Wih1, Wih1b, 4 * H_DIM * H_DIM);
  cvt(Whh1, Whh1b, 4 * H_DIM * H_DIM);
  cvt(h0,   hbf,   2 * B_DIM * H_DIM);
  hipMemcpyAsync(hst, h0, (size_t)2 * B_DIM * H_DIM * 4, hipMemcpyDeviceToDevice, stream);
  hipMemcpyAsync(cst, c0, (size_t)2 * B_DIM * H_DIM * 4, hipMemcpyDeviceToDevice, stream);

  // 2) MLP: two big WMMA GEMMs (wave tile 32x64, block 64x256)
  {
    dim3 grid(H_DIM / 256, TB_DIM / 64);
    wmma_gemm_kernel<float, 2, 4, true, false, true><<<grid, 256, 0, stream>>>(
        obs, W1b, OBS_DIM, nullptr, nullptr, 0, nullptr, b1, nullptr,
        nullptr, X1b, H_DIM);
    wmma_gemm_kernel<bf16_t, 2, 4, true, false, true><<<grid, 256, 0, stream>>>(
        X1b, W2b, H_DIM, nullptr, nullptr, 0, nullptr, b2, nullptr,
        nullptr, CIb, H_DIM);
  }

  // 3) LSTM scan: per step, per layer: fused gates GEMM (x@Wih^T + nd*h@Whh^T
  //    + bih + bhh) then elementwise cell update.  Wave tile 16x32 spreads the
  //    small 64x4096 GEMM over 64 workgroups.
  {
    dim3 ggrid((4 * H_DIM) / 128, B_DIM / 32);
    const int cblocks = (B_DIM * H_DIM + 255) / 256;
    bf16_t* h0b = hbf;
    bf16_t* h1b = hbf + (size_t)B_DIM * H_DIM;
    float *h0s = hst, *h1s = hst + (size_t)B_DIM * H_DIM;
    float *c0s = cst, *c1s = cst + (size_t)B_DIM * H_DIM;
    for (int t = 0; t < T_DIM; t++) {
      const bf16_t* xt = CIb + (size_t)t * B_DIM * H_DIM;
      const int* dt = done + (size_t)t * B_DIM;
      // layer 0
      wmma_gemm_kernel<bf16_t, 1, 2, false, true, false><<<ggrid, 256, 0, stream>>>(
          xt, Wih0b, H_DIM, h0b, Whh0b, H_DIM, dt, bih0, bhh0,
          gates, nullptr, 4 * H_DIM);
      lstm_cell_kernel<<<cblocks, 256, 0, stream>>>(gates, dt, h0s, c0s, h0b, nullptr);
      // layer 1 (x input = fresh h0, no mask on it; mask on old h1)
      wmma_gemm_kernel<bf16_t, 1, 2, false, true, false><<<ggrid, 256, 0, stream>>>(
          h0b, Wih1b, H_DIM, h1b, Whh1b, H_DIM, dt, bih1, bhh1,
          gates, nullptr, 4 * H_DIM);
      lstm_cell_kernel<<<cblocks, 256, 0, stream>>>(
          gates, dt, h1s, c1s, h1b, COb + (size_t)t * B_DIM * H_DIM);
    }
  }

  // 4) Heads + argmax.  d_out layout: logits | baseline | action | h_f | c_f
  {
    float* logits   = out;
    float* baseline = out + (size_t)TB_DIM * A_DIM;
    float* action   = out + (size_t)TB_DIM * A_DIM + TB_DIM;
    heads_kernel<<<TB_DIM / 8, 256, 0, stream>>>(COb, Wp, bp, Wb, bb,
                                                 logits, baseline, action, TB_DIM);
    float* hf = out + (size_t)TB_DIM * A_DIM + 2 * TB_DIM;
    float* cf = hf + (size_t)2 * B_DIM * H_DIM;
    hipMemcpyAsync(hf, hst, (size_t)2 * B_DIM * H_DIM * 4, hipMemcpyDeviceToDevice, stream);
    hipMemcpyAsync(cf, cst, (size_t)2 * B_DIM * H_DIM * 4, hipMemcpyDeviceToDevice, stream);
  }
}
